// TemporalAugmentor_77292231459404
// MI455X (gfx1250) — matrix operand order repaired and checked by reading, not compiled, not run on hardware
//
#include <hip/hip_runtime.h>
#include <hip/hip_bf16.h>

// ---------------------------------------------------------------------------
// LSTM recurrence on gfx1250: weights resident in WGP LDS (320KB), gate GEMM
// via chained v_wmma_f32_16x16x32_bf16 (fp32 accumulate, fp32 cell state).
// Grid: 64 WGs x 1024 threads (32 waves); each WG owns 16 batch rows.
// Biases are folded into the WMMA accumulator init (column-splat); the
// teacher-phase output projections are deferred to a parallel post-pass
// (h history staged in d_ws) so the 127-step serial chain only carries
// gates -> cell-update; autoregressive steps keep the inline y (feedback).
// Activations use v_exp_f32 + v_rcp_f32 (no IEEE division on the chain).
// ---------------------------------------------------------------------------

typedef __attribute__((ext_vector_type(16))) __bf16        v16bf;
typedef __attribute__((ext_vector_type(8)))  float         v8f;
typedef __attribute__((ext_vector_type(4)))  unsigned int  v4u;
typedef __attribute__((ext_vector_type(8)))  unsigned int  v8u;

namespace {
constexpr int Bb  = 1024;   // batch
constexpr int Tt  = 128;    // timesteps
constexpr int NBc = 13;     // feature dim
constexpr int Hc  = 128;    // hidden
constexpr int TIN = 64;     // teacher-forced input steps
constexpr int MT  = 16;     // batch rows per workgroup

constexpr int KP = 168;     // padded K for [x|h] (13+128 -> 168 bf16); 336B rows
constexpr int HP = 136;     // padded H for h / W_out rows; 272B rows
constexpr int GS = 516;     // gates row stride in floats (bank-spread)

constexpr int OFF_WCAT = 0;                         // [512][KP] bf16 = 172032 B
constexpr int OFF_A    = OFF_WCAT + 512 * KP * 2;   // [16][KP]  bf16
constexpr int OFF_H    = OFF_A    + MT  * KP * 2;   // [16][HP]  bf16
constexpr int OFF_WO   = OFF_H    + MT  * HP * 2;   // [16][HP]  bf16
constexpr int OFF_G    = OFF_WO   + MT  * HP * 2;   // [16][GS]  f32
constexpr int OFF_BG   = OFF_G    + MT  * GS * 4;   // [512]     f32
constexpr int OFF_BO   = OFF_BG   + 512 * 4;        // [16]      f32
constexpr int SMEM_BYTES = OFF_BO + 64;             // ~221 KB < 320 KB WGP LDS
static_assert(SMEM_BYTES <= 320 * 1024, "fits in WGP LDS");

constexpr size_t WS_NEEDED = (size_t)Bb * TIN * Hc * 2;  // bf16 h history, 16 MB
}

__device__ __forceinline__ unsigned short f2bf(float f) {
  unsigned int u = __builtin_bit_cast(unsigned int, f);
  u += 0x7FFFu + ((u >> 16) & 1u);          // round-to-nearest-even
  return (unsigned short)(u >> 16);
}
// v_exp_f32 + v_rcp_f32; ~1 ulp, no v_div_scale/div_fmas sequences.
__device__ __forceinline__ float fast_sigmoid(float x) {
  return __builtin_amdgcn_rcpf(1.0f + __expf(-x));
}
__device__ __forceinline__ float fast_tanh(float x) {
  return 1.0f - 2.0f * __builtin_amdgcn_rcpf(1.0f + __expf(2.0f * x));
}

__device__ __forceinline__ v8f wmma_bf16(v8u a, v8u b, v8f c) {
  return __builtin_amdgcn_wmma_f32_16x16x32_bf16(
      false, __builtin_bit_cast(v16bf, a),
      false, __builtin_bit_cast(v16bf, b),
      (short)0, c, false, false);
}
__device__ __forceinline__ v8f splat8(float v) {
  v8f r = {v, v, v, v, v, v, v, v};
  return r;
}

__global__ __launch_bounds__(1024)
void lstm_wmma_kernel(const float* __restrict__ x,
                      const float* __restrict__ Wih,
                      const float* __restrict__ Whh,
                      const float* __restrict__ bih,
                      const float* __restrict__ bhh,
                      const float* __restrict__ Wout,
                      const float* __restrict__ bout,
                      float* __restrict__ out,
                      unsigned short* __restrict__ hws,
                      int defer) {
  extern __shared__ unsigned char smem[];
  unsigned short* sWcat = (unsigned short*)(smem + OFF_WCAT); // [512][KP] rows of [W_ih|W_hh]
  unsigned short* sA    = (unsigned short*)(smem + OFF_A);    // [16][KP]  [x_t | h] bf16
  unsigned short* sH    = (unsigned short*)(smem + OFF_H);    // [16][HP]  h bf16 (aligned copy)
  unsigned short* sWo   = (unsigned short*)(smem + OFF_WO);   // [16][HP]  W_out rows (pad->16)
  float*          sG    = (float*)(smem + OFF_G);             // [16][GS]  gates f32
  float*          sBg   = (float*)(smem + OFF_BG);            // [512] b_ih+b_hh
  float*          sBo   = (float*)(smem + OFF_BO);            // [16]  b_out (pad)

  const int tid  = threadIdx.x;
  const int lane = tid & 31;
  const int wid  = tid >> 5;
  const int b0   = blockIdx.x * MT;

  // ---- stage weights into LDS as bf16 (one time; served from 192MB L2) ----
  for (int idx = tid; idx < 512 * KP; idx += 1024) {
    int n = idx / KP, k = idx % KP;
    float v = 0.0f;
    if (k < NBc)           v = Wih[n * NBc + k];
    else if (k < NBc + Hc) v = Whh[n * Hc + (k - NBc)];
    sWcat[idx] = f2bf(v);
  }
  for (int idx = tid; idx < MT * HP; idx += 1024) {
    int j = idx / HP, k = idx % HP;
    float v = (j < NBc && k < Hc) ? Wout[j * Hc + k] : 0.0f;
    sWo[idx] = f2bf(v);
  }
  if (tid < 512) sBg[tid] = bih[tid] + bhh[tid];
  if (tid < 16)  sBo[tid] = (tid < NBc) ? bout[tid] : 0.0f;
  for (int idx = tid; idx < MT * KP; idx += 1024) sA[idx] = 0;  // h0 = 0, pad = 0
  __syncthreads();

  // ---- out[:,0] = x[:,0]; seed A with x[:,0] ----
  if (tid < MT * NBc) {
    int m = tid / NBc, j = tid % NBc;
    float xv = x[((size_t)(b0 + m) * Tt + 0) * NBc + j];
    sA[m * KP + j] = f2bf(xv);
    out[((size_t)(b0 + m) * Tt + 0) * NBc + j] = xv;
  }
  __syncthreads();

  const int n0     = wid * 16;            // gate-column tile for this wave
  const int mlo    = lane & 15;
  const int khalfA = (lane >> 4) << 3;    // A frag: +0 / +8  (ISA 16-bit A layout)
  const int khalfB = (lane >> 4) << 4;    // unused; B frags load with khalfA (+0 / +8) like A
  const int mrow   = (lane >> 4) << 3;    // C/D frag row base: 0 / 8
  float creg0 = 0.0f, creg1 = 0.0f;       // fp32 cell state, 2 elems/thread

  // Loop-invariant bias splats: gate bias depends only on the output column,
  // and all 8 C/D fragment elements of a lane share one column.
  const float biasG = sBg[n0 + mlo];      // gate-GEMM accumulator init
  const float biasY = sBo[mlo];           // y-GEMM accumulator init

  for (int t = 0; t < Tt - 1; ++t) {
    // ---- gates[16 x 512] = [x|h] @ [W_ih|W_hh]^T + bias ----
    // Preload all 5 A/B fragment pairs, then run the WMMA chain back-to-back.
    const unsigned short* arow = sA    + mlo * KP;
    const unsigned short* brow = sWcat + (size_t)(n0 + mlo) * KP;
    v8u afr[5], bfr[5];
#pragma unroll
    for (int i = 0; i < 5; ++i) {
      const int k0 = i * 32;
      v4u a_lo = *(const v4u*)(arow + k0 + khalfA);
      v4u a_hi = *(const v4u*)(arow + k0 + 16 + khalfA);
      v4u b_lo = *(const v4u*)(brow + k0 + khalfA);
      v4u b_hi = *(const v4u*)(brow + k0 + 16 + khalfA);
      afr[i].lo = a_lo; afr[i].hi = a_hi;
      bfr[i].lo = b_lo; bfr[i].hi = b_hi;
    }
    v8f acc = splat8(biasG);
#pragma unroll
    for (int i = 0; i < 5; ++i) acc = wmma_bf16(afr[i], bfr[i], acc);
    {
      const int nc = n0 + mlo;
#pragma unroll
      for (int r = 0; r < 8; ++r) sG[(mrow + r) * GS + nc] = acc[r];
    }
    __syncthreads();

    // ---- fused cell update: i,f,g,o -> c,h (2 elements per thread) ----
#pragma unroll
    for (int p = 0; p < 2; ++p) {
      int e  = tid + p * 1024;
      int m  = e >> 7, hc = e & 127;
      const float* gr = sG + m * GS;
      float iv = fast_sigmoid(gr[hc]);
      float fv = fast_sigmoid(gr[128 + hc]);
      float gv = fast_tanh   (gr[256 + hc]);
      float ov = fast_sigmoid(gr[384 + hc]);
      float cp = p ? creg1 : creg0;
      float cn = fv * cp + iv * gv;
      if (p) creg1 = cn; else creg0 = cn;
      unsigned short hb = f2bf(ov * fast_tanh(cn));
      sA[m * KP + NBc + hc] = hb;   // h part of next-step A fragment
      sH[m * HP + hc]       = hb;   // aligned h copy for the inline y-GEMM
      if (defer && t < TIN)         // stage h history for the deferred y pass
        hws[((size_t)(b0 + m) * TIN + t) * Hc + hc] = hb;
    }
    // teacher-forced input for next step (steps 1..64 incl. the seed x[:,64])
    if (t + 1 <= TIN && tid < MT * NBc) {
      int m = tid / NBc, j = tid % NBc;
      sA[m * KP + j] = f2bf(x[((size_t)(b0 + m) * Tt + (t + 1)) * NBc + j]);
    }
    __syncthreads();

    // ---- inline y = h @ W_out^T + b_out (wave 0) ----
    // Runs every step when not deferring; only for autoregressive steps
    // (where y is the next input) when deferring.
    if (!defer || t >= TIN) {
      if (wid == 0) {
        const unsigned short* hrow = sH  + mlo * HP;
        const unsigned short* wrow = sWo + mlo * HP;
        v8u ha[4], wb[4];
#pragma unroll
        for (int i = 0; i < 4; ++i) {
          const int k0 = i * 32;
          v4u a_lo = *(const v4u*)(hrow + k0 + khalfA);
          v4u a_hi = *(const v4u*)(hrow + k0 + 16 + khalfA);
          v4u b_lo = *(const v4u*)(wrow + k0 + khalfA);
          v4u b_hi = *(const v4u*)(wrow + k0 + 16 + khalfA);
          ha[i].lo = a_lo; ha[i].hi = a_hi;
          wb[i].lo = b_lo; wb[i].hi = b_hi;
        }
        v8f yacc = splat8(biasY);
#pragma unroll
        for (int i = 0; i < 4; ++i) yacc = wmma_bf16(ha[i], wb[i], yacc);
        const int j = mlo;
        if (j < NBc) {
#pragma unroll
          for (int r = 0; r < 8; ++r) {
            float yv = yacc[r];
            int   m  = mrow + r;
            out[((size_t)(b0 + m) * Tt + (t + 1)) * NBc + j] = yv;
            if (t >= TIN) sA[m * KP + j] = f2bf(yv);   // autoregressive feedback
          }
        }
      }
      __syncthreads();
    }
  }

  // ---- deferred teacher-phase projections: ys1 = h_t @ W_out^T + b_out ----
  // 64 independent 16x16 tiles (one per teacher step) across 32 waves.
  if (defer) {
    __threadfence();   // order h-history stores before cross-lane loads
    __syncthreads();
    for (int tt = wid; tt < TIN; tt += 32) {
      const unsigned short* hrow = hws + ((size_t)(b0 + mlo) * TIN + tt) * Hc;
      const unsigned short* wrow = sWo + mlo * HP;
      v8u ha[4], wb[4];
#pragma unroll
      for (int i = 0; i < 4; ++i) {
        const int k0 = i * 32;
        v4u a_lo = *(const v4u*)(hrow + k0 + khalfA);      // global_load_b128
        v4u a_hi = *(const v4u*)(hrow + k0 + 16 + khalfA);
        v4u b_lo = *(const v4u*)(wrow + k0 + khalfA);
        v4u b_hi = *(const v4u*)(wrow + k0 + 16 + khalfA);
        ha[i].lo = a_lo; ha[i].hi = a_hi;
        wb[i].lo = b_lo; wb[i].hi = b_hi;
      }
      v8f yacc = splat8(biasY);
#pragma unroll
      for (int i = 0; i < 4; ++i) yacc = wmma_bf16(ha[i], wb[i], yacc);
      const int j = mlo;
      if (j < NBc) {
#pragma unroll
        for (int r = 0; r < 8; ++r) {
          int m = mrow + r;
          out[((size_t)(b0 + m) * Tt + (tt + 1)) * NBc + j] = yacc[r];
        }
      }
    }
  }
}

extern "C" void kernel_launch(void* const* d_in, const int* in_sizes, int n_in,
                              void* d_out, int out_size, void* d_ws, size_t ws_size,
                              hipStream_t stream) {
  (void)in_sizes; (void)n_in; (void)out_size;
  const float* x    = (const float*)d_in[0];
  const float* Wih  = (const float*)d_in[1];
  const float* Whh  = (const float*)d_in[2];
  const float* bih  = (const float*)d_in[3];
  const float* bhh  = (const float*)d_in[4];
  const float* Wout = (const float*)d_in[5];
  const float* bout = (const float*)d_in[6];
  float* out = (float*)d_out;

  const int defer = (d_ws != nullptr && ws_size >= WS_NEEDED) ? 1 : 0;

  // Allow >64KB dynamic LDS (gfx1250 WGP has 320KB). Idempotent, capture-safe.
  hipFuncSetAttribute(reinterpret_cast<const void*>(lstm_wmma_kernel),
                      hipFuncAttributeMaxDynamicSharedMemorySize, SMEM_BYTES);

  lstm_wmma_kernel<<<dim3(Bb / MT), dim3(1024), SMEM_BYTES, stream>>>(
      x, Wih, Whh, bih, bhh, Wout, bout, out,
      (unsigned short*)d_ws, defer);
}
